// Generator_32427003085030
// MI455X (gfx1250) — matrix operand order repaired and checked by reading, not compiled, not run on hardware
//
#include <hip/hip_runtime.h>
#include <hip/hip_bf16.h>
#include <math.h>

// ---------------------------------------------------------------------------
// HouseGAN-like generator on MI455X (gfx1250), wave32 + WMMA implicit GEMM.
//
// All activations are NHWC _Float16. Convs are implicit GEMMs on
// v_wmma_f32_16x16x32_f16: D(16 Cout x 16 pixels) = sum over 9 taps and
// Cin/32 K-slices of A(Cout x 32ci) * B(32ci x 16pix), f32 accumulate.
//
// Fragment layouts used here (same contiguous k order in A and B):
//   A 16x32 f16 : lane -> M = lane&15 ; element e: K = hi*16+e
//                 (hi = lane>>4)  => two contiguous 8-half
//                 runs in a [row][K] (K-contiguous) weight layout.
//   B 32x16 f16 : lane -> N = lane&15 ; element e: K = hi*16+e
//                 => 16 contiguous halfs in a [pixel][ci] LDS tile.
//   C/D f32     : lane -> N = lane&15 ; reg r -> M = r + 8*hi  => 8
//                 consecutive Cout per lane => one packed v8h store in NHWC.
//
// LDS staging for non-upsampling convs uses the CDNA5 async copy path
// (global_load_async_to_lds_b128 + s_wait_asynccnt), one 16-byte chunk per
// lane per op: global NHWC rows and the LDS [pixel][ci] tile are both
// contiguous per image row, so the interior of the halo tile is a straight
// row-segment DMA.
// ---------------------------------------------------------------------------

#define USE_ASYNC_STAGING 1

typedef _Float16 h16;
typedef __attribute__((ext_vector_type(16))) _Float16 v16h;
typedef __attribute__((ext_vector_type(8)))  _Float16 v8h;
typedef __attribute__((ext_vector_type(8)))  float    v8f;

#define NROOMS 1024
#define NEDGES 32768

__device__ __forceinline__ float lrelu_f(float v) { return v >= 0.f ? v : 0.2f * v; }

#if USE_ASYNC_STAGING
// Async global -> LDS 16-byte copy (per active lane). VDST VGPR carries the
// LDS byte offset (flat LDS aperture: addr[31:0] == LDS offset).
__device__ __forceinline__ void async_g2l_b128(unsigned lds_off, const h16* g)
{
    asm volatile("global_load_async_to_lds_b128 %0, %1, off"
                 :: "v"(lds_off), "v"(g) : "memory");
}
__device__ __forceinline__ void wait_async0()
{
    asm volatile("s_wait_asynccnt 0" ::: "memory");
}
#endif

// ---------------------------------------------------------------------------
// Generic 3x3 SAME conv, NHWC f16 in/out, WMMA implicit GEMM.
//  CIN  : input channels (multiple of 32)
//  COUT : output channels, padded to multiple of 16
//  H,W  : OUTPUT spatial dims. UPS=1 => input is (H/2,W/2), nearest-up2 fused.
//  SH   : strip rows per block (LDS = (SH+2)*(W+2)*CIN halfs <= 64KB)
//  EPI  : 0 = store raw f16 (pre-instance-norm)
//         1 = lrelu, optional f16 residual add, store f16
//         2 = sigmoid on channel 0 only, store f32 (final layer)
// ---------------------------------------------------------------------------
template <int CIN, int COUT, int H, int W, int SH, int UPS, int EPI>
__global__ __launch_bounds__(256) void conv3x3_wmma_k(
    const h16* __restrict__ in, const h16* __restrict__ wpack,
    const float* __restrict__ bias, h16* __restrict__ out,
    const h16* __restrict__ res, float* __restrict__ out32)
{
    constexpr int Wp   = W + 2;
    constexpr int SRCH = UPS ? H / 2 : H;
    constexpr int SRCW = UPS ? W / 2 : W;
    constexpr int NPIX = SH * W;
    constexpr int NPT  = NPIX / 16;   // pixel tiles per strip
    constexpr int NMT  = COUT / 16;   // Cout tiles
    constexpr int TT   = NMT * NPT;
    constexpr int TOT  = (SH + 2) * Wp * CIN;

    __shared__ h16 smem[TOT];

    const int img = blockIdx.x;
    const int y0  = blockIdx.y * SH;
    const int tid = threadIdx.x;

    // hint the weight pack into cache (global_prefetch_b8)
    __builtin_prefetch(wpack + (size_t)tid * 64, 0, 3);

    const size_t inBase = (size_t)img * SRCH * SRCW * CIN;

#if USE_ASYNC_STAGING
    if constexpr (UPS == 0) {
        // -- zero the whole halo tile with vector LDS stores --
        v8h zv;
        #pragma unroll
        for (int i = 0; i < 8; ++i) zv[i] = (h16)0.f;
        for (int e = tid * 8; e < TOT; e += 256 * 8)
            *(v8h*)(smem + e) = zv;
        __syncthreads();   // compiler inserts s_wait_dscnt before barrier

        // -- async DMA of interior rows: global NHWC row -> LDS [pix][ci] --
        const unsigned smemBase = (unsigned)(uintptr_t)smem;  // LDS byte offset
        constexpr int CHUNKS = (W * CIN * 2) / 16;            // 16B per lane-op
        for (int rr = 0; rr < SH + 2; ++rr) {
            const int py = y0 - 1 + rr;
            if (py < 0 || py >= H) continue;
            const unsigned ldsRow = smemBase + (unsigned)((rr * Wp + 1) * CIN * 2);
            const h16* gRow = in + inBase + (size_t)py * W * CIN;
            for (int ch = tid; ch < CHUNKS; ch += 256)
                async_g2l_b128(ldsRow + ch * 16, gRow + ch * 8);
        }
        wait_async0();
        __syncthreads();
    } else
#endif
    {
        // ---- elementwise staging (handles fused nearest-neighbor up2) ----
        for (int e = tid; e < TOT; e += 256) {
            const int c  = e % CIN;
            const int pq = e / CIN;
            const int px = pq % Wp - 1;
            const int py = pq / Wp + y0 - 1;
            h16 v = (h16)0.f;
            if (py >= 0 && py < H && px >= 0 && px < W) {
                const int sy = UPS ? (py >> 1) : py;
                const int sx = UPS ? (px >> 1) : px;
                v = in[inBase + ((size_t)sy * SRCW + sx) * CIN + c];
            }
            smem[e] = v;
        }
        __syncthreads();
    }

    const int wv   = tid >> 5;
    const int lane = tid & 31;
    const int li   = lane & 15;
    const int hi   = lane >> 4;

    for (int t = wv; t < TT; t += 8) {       // wave-uniform => EXEC all-ones
        const int mt  = t / NPT;
        const int p0  = (t % NPT) * 16;
        const int pix = p0 + li;             // strip-local pixel for B / D
        const int ly  = pix / W, lx = pix % W;

        v8f acc;
        #pragma unroll
        for (int r = 0; r < 8; ++r) {
            const int co = mt * 16 + hi * 8 + r;
            float bv;
            if constexpr (EPI == 2) bv = (co == 0) ? bias[0] : 0.f;  // padded Cout
            else                    bv = bias[co];
            acc[r] = bv;
        }

        #pragma unroll
        for (int dydx = 0; dydx < 9; ++dydx) {
            const int dy = dydx / 3, dx = dydx % 3;
            const h16* bptr = smem + ((ly + dy) * Wp + (lx + dx)) * CIN + hi * 16;
            const h16* aptr = wpack + ((size_t)dydx * COUT + mt * 16 + li) * CIN + hi * 16;
            #pragma unroll
            for (int ks = 0; ks < CIN / 32; ++ks) {
                const v16h bfrag = *(const v16h*)(bptr + ks * 32);       // ds b128 x2
                const v8h  a0    = *(const v8h*)(aptr + ks * 32);        // global b128
                const v8h  a1    = *(const v8h*)(aptr + ks * 32 + 8);
                v16h afrag;
                #pragma unroll
                for (int i = 0; i < 8; ++i) { afrag[i] = a0[i]; afrag[i + 8] = a1[i]; }
                acc = __builtin_amdgcn_wmma_f32_16x16x32_f16(
                    false, afrag, false, bfrag, (short)0, acc, false, false);
            }
        }

        const int gpix = (y0 + ly) * W + lx;
        if constexpr (EPI == 2) {
            if (mt == 0 && hi == 0) {                    // channel 0 only
                const float v = acc[0];
                out32[(size_t)img * H * W + gpix] = 1.f / (1.f + __expf(-v));
            }
        } else {
            const size_t obase = ((size_t)img * H * W + gpix) * COUT + mt * 16 + hi * 8;
            v8h o;
            #pragma unroll
            for (int r = 0; r < 8; ++r) {
                float v = acc[r];
                if constexpr (EPI == 1) v = lrelu_f(v);
                if (res) v += (float)res[obase + r];
                o[r] = (h16)v;
            }
            *(v8h*)(out + obase) = o;                    // packed b128 store
        }
    }
}

// ---------------------------------------------------------------------------
// Linear layer: init_feat = zy(1024x160) @ l1_w(160x2048) + b, as WMMA GEMM.
// A = W^T [feat][k] (pre-packed f16), B = zy [room][k]. D rows=feat, cols=room.
// Stored directly into NHWC init-feature buffer with feat = c*64 + p.
// ---------------------------------------------------------------------------
__global__ __launch_bounds__(256) void lin_gemm_k(
    const h16* __restrict__ wT, const h16* __restrict__ zy,
    const float* __restrict__ bias, h16* __restrict__ outf)
{
    constexpr int K = 160;
    const int wv = threadIdx.x >> 5, lane = threadIdx.x & 31;
    const int li = lane & 15, hi = lane >> 4;
    const int tile = blockIdx.x * 8 + wv;     // 8192 tiles total
    const int mt = tile / 64;                 // feat tile 0..127
    const int nt = tile % 64;                 // room tile 0..63
    const int feat0 = mt * 16;
    const int room  = nt * 16 + li;

    v8f acc;
    #pragma unroll
    for (int r = 0; r < 8; ++r) acc[r] = bias[feat0 + hi * 8 + r];

    const h16* aptr = wT + (size_t)(feat0 + li) * K + hi * 16;
    const h16* bptr = zy + (size_t)room * K + hi * 16;
    #pragma unroll
    for (int ks = 0; ks < K / 32; ++ks) {
        const v16h bfrag = *(const v16h*)(bptr + ks * 32);
        const v8h  a0    = *(const v8h*)(aptr + ks * 32);
        const v8h  a1    = *(const v8h*)(aptr + ks * 32 + 8);
        v16h afrag;
        #pragma unroll
        for (int i = 0; i < 8; ++i) { afrag[i] = a0[i]; afrag[i + 8] = a1[i]; }
        acc = __builtin_amdgcn_wmma_f32_16x16x32_f16(
            false, afrag, false, bfrag, (short)0, acc, false, false);
    }
    #pragma unroll
    for (int r = 0; r < 8; ++r) {
        const int feat = feat0 + hi * 8 + r;
        const int c = feat >> 6, p = feat & 63;   // reshape(n, C, 8, 8)
        outf[((size_t)room * 64 + p) * 32 + c] = (h16)acc[r];
    }
}

// ---------------------------------------------------------------------------
// Small helper kernels
// ---------------------------------------------------------------------------
// zy = concat(z[nd[room]], given_y[room]), zero-padded 146 -> 160, f16.
__global__ void zy_k(const float* __restrict__ z, const float* __restrict__ gy,
                     const int* __restrict__ nd, h16* __restrict__ zy)
{
    const int idx = blockIdx.x * blockDim.x + threadIdx.x;
    if (idx >= NROOMS * 160) return;
    const int room = idx / 160, k = idx % 160;
    float v = 0.f;
    if (k < 128)      v = z[(size_t)nd[room] * 128 + k];
    else if (k < 146) v = gy[(size_t)room * 18 + (k - 128)];
    zy[idx] = (h16)v;
}

// l1_w (146x2048 row-major) -> W^T f16 [2048][160] zero-padded.
__global__ void packlin_k(const float* __restrict__ w, h16* __restrict__ o)
{
    const int idx = blockIdx.x * blockDim.x + threadIdx.x;
    if (idx >= 2048 * 160) return;
    const int f = idx / 160, k = idx % 160;
    o[idx] = (h16)((k < 146) ? w[(size_t)k * 2048 + f] : 0.f);
}

// conv weights OIHW f32 -> [9][CoutP][Cin] f16 (Cout zero-padded to CoutP).
__global__ void packconv_k(const float* __restrict__ w, h16* __restrict__ o,
                           int Cout, int CoutP, int Cin)
{
    const int idx = blockIdx.x * blockDim.x + threadIdx.x;
    const int total = 9 * CoutP * Cin;
    if (idx >= total) return;
    const int dydx = idx / (CoutP * Cin);
    const int rem  = idx % (CoutP * Cin);
    const int co = rem / Cin, ci = rem % Cin;
    o[idx] = (h16)((co < Cout) ? w[((size_t)co * Cin + ci) * 9 + dydx] : 0.f);
}

// enc conv1: 1 -> 64 channels at 8x8 (K=9, too small for WMMA) -> f16 NHWC.
__global__ void enc1a_k(const float* __restrict__ gm, const float* __restrict__ w,
                        const float* __restrict__ b, h16* __restrict__ e1)
{
    const int idx = blockIdx.x * blockDim.x + threadIdx.x;   // (img, p, co)
    if (idx >= NROOMS * 64 * 64) return;
    const int co = idx % 64, p = (idx / 64) % 64, img = idx / (64 * 64);
    const int y = p / 8, x = p % 8;
    float s = b[co];
    #pragma unroll
    for (int dy = 0; dy < 3; ++dy)
        #pragma unroll
        for (int dx = 0; dx < 3; ++dx) {
            const int yy = y + dy - 1, xx = x + dx - 1;
            if (yy >= 0 && yy < 8 && xx >= 0 && xx < 8)
                s += gm[(size_t)img * 64 + yy * 8 + xx] * w[co * 9 + dy * 3 + dx];
        }
    e1[idx] = (h16)lrelu_f(s);
}

// GNN segment-sum: one block per edge, f32 atomics into pos/neg pools.
__global__ __launch_bounds__(256) void edge_k(const int* __restrict__ gw,
                                              const h16* __restrict__ x,
                                              float* __restrict__ pos,
                                              float* __restrict__ neg)
{
    const int e = blockIdx.x;
    const int src = gw[e * 3 + 0], typ = gw[e * 3 + 1], tgt = gw[e * 3 + 2];
    if (src < 0 || src >= NROOMS || tgt < 0 || tgt >= NROOMS || typ == 0) return;
    float* dst = (typ > 0 ? pos : neg) + (size_t)tgt * 2048;
    const h16* s = x + (size_t)src * 2048;
    for (int i = threadIdx.x; i < 2048; i += 256)
        atomicAdd(dst + i, (float)s[i]);
}

// comb = concat(x, pos, neg) channel-wise, NHWC 96-ch f16.
__global__ void comb_k(const h16* __restrict__ x, const float* __restrict__ pos,
                       const float* __restrict__ neg, h16* __restrict__ comb)
{
    const int idx = blockIdx.x * blockDim.x + threadIdx.x;   // (room, p, c96)
    if (idx >= NROOMS * 64 * 96) return;
    const int c = idx % 96, p = (idx / 96) % 64, room = idx / (96 * 64);
    float v;
    if (c < 32)      v = (float)x[((size_t)room * 64 + p) * 32 + c];
    else if (c < 64) v = pos[(size_t)room * 2048 + p * 32 + (c - 32)];
    else             v = neg[(size_t)room * 2048 + p * 32 + (c - 64)];
    comb[idx] = (h16)v;
}

// x += 0.01 * noise (noise NCHW f32 -> NHWC f16 in-place).
__global__ void noise_k(h16* __restrict__ x, const float* __restrict__ nz)
{
    const int idx = blockIdx.x * blockDim.x + threadIdx.x;   // (room, p, c)
    if (idx >= NROOMS * 64 * 32) return;
    const int c = idx % 32, p = (idx / 32) % 64, room = idx / (32 * 64);
    x[idx] = (h16)((float)x[idx] + 0.01f * nz[((size_t)room * 32 + c) * 64 + p]);
}

// instance norm + lrelu, in place. One thread per channel, coalesced in NHWC.
template <int C, int P>
__global__ void inorm_k(h16* __restrict__ a)
{
    const int img = blockIdx.x, t = threadIdx.x;
    const size_t base = (size_t)img * P * C + t;
    float s = 0.f, ss = 0.f;
    for (int p = 0; p < P; ++p) {
        const float v = (float)a[base + (size_t)p * C];
        s += v; ss += v * v;
    }
    const float mu  = s / P;
    const float var = ss / P - mu * mu;
    const float inv = rsqrtf(var + 1e-5f);
    for (int p = 0; p < P; ++p) {
        const float v = ((float)a[base + (size_t)p * C] - mu) * inv;
        a[base + (size_t)p * C] = (h16)lrelu_f(v);
    }
}

// ---------------------------------------------------------------------------
// Host launch
// ---------------------------------------------------------------------------
extern "C" void kernel_launch(void* const* d_in, const int* in_sizes, int n_in,
                              void* d_out, int out_size, void* d_ws, size_t ws_size,
                              hipStream_t stream)
{
    // setup_inputs() dict order; params dict in insertion order.
    const float* z    = (const float*)d_in[0];
    const float* gm   = (const float*)d_in[1];
    const float* gy   = (const float*)d_in[2];
    const float* nz   = (const float*)d_in[3];
    const float* l1_w = (const float*)d_in[4];
    const float* l1_b = (const float*)d_in[5];
    const float* e_w1 = (const float*)d_in[6];
    const float* e_b1 = (const float*)d_in[7];
    const float* e_w2 = (const float*)d_in[8];
    const float* e_b2 = (const float*)d_in[9];
    const float* cw1[3], *cb1[3], *cw2[3], *cb2[3];
    for (int i = 0; i < 3; ++i) {
        cw1[i] = (const float*)d_in[10 + 4 * i];
        cb1[i] = (const float*)d_in[11 + 4 * i];
        cw2[i] = (const float*)d_in[12 + 4 * i];
        cb2[i] = (const float*)d_in[13 + 4 * i];
    }
    const float* up1_w = (const float*)d_in[22];
    const float* up1_b = (const float*)d_in[23];
    const float* up2_w = (const float*)d_in[24];
    const float* up2_b = (const float*)d_in[25];
    const float* d1_w  = (const float*)d_in[26];
    const float* d1_b  = (const float*)d_in[27];
    const float* d2_w  = (const float*)d_in[28];
    const float* d2_b  = (const float*)d_in[29];
    const float* d3_w  = (const float*)d_in[30];
    const float* d3_b  = (const float*)d_in[31];
    const int*   gw    = (const int*)d_in[32];
    const int*   nd    = (const int*)d_in[33];
    float* outp = (float*)d_out;

    // ---- carve workspace ----
    size_t off = 0;
    auto take = [&](size_t bytes) -> void* {
        void* p = (char*)d_ws + off;
        off += (bytes + 255) & ~(size_t)255;
        return p;
    };
    h16* zy     = (h16*)take((size_t)NROOMS * 160 * 2);
    h16* wlinT  = (h16*)take((size_t)2048 * 160 * 2);
    h16* initf  = (h16*)take((size_t)NROOMS * 64 * 32 * 2);
    h16* e1     = (h16*)take((size_t)NROOMS * 64 * 64 * 2);
    h16* x8     = (h16*)take((size_t)NROOMS * 64 * 32 * 2);
    h16* h1     = (h16*)take((size_t)NROOMS * 64 * 64 * 2);
    h16* comb   = (h16*)take((size_t)NROOMS * 64 * 96 * 2);
    float* pos  = (float*)take((size_t)NROOMS * 2048 * 4);
    float* neg  = (float*)take((size_t)NROOMS * 2048 * 4);
    h16* wpEnc  = (h16*)take((size_t)9 * 32 * 64 * 2);
    h16* wpC1[3], *wpC2[3];
    for (int i = 0; i < 3; ++i) {
        wpC1[i] = (h16*)take((size_t)9 * 64 * 96 * 2);
        wpC2[i] = (h16*)take((size_t)9 * 32 * 64 * 2);
    }
    h16* wpU1 = (h16*)take((size_t)9 * 64 * 32 * 2);
    h16* wpU2 = (h16*)take((size_t)9 * 128 * 64 * 2);
    h16* wpD1 = (h16*)take((size_t)9 * 64 * 128 * 2);
    h16* wpD2 = (h16*)take((size_t)9 * 32 * 64 * 2);
    h16* wpD3 = (h16*)take((size_t)9 * 16 * 32 * 2);
    h16* a16  = (h16*)take((size_t)NROOMS * 256 * 64 * 2);
    h16* a32a = (h16*)take((size_t)NROOMS * 1024 * 128 * 2);
    h16* a32b = (h16*)take((size_t)NROOMS * 1024 * 64 * 2);
    h16* a32c = (h16*)take((size_t)NROOMS * 1024 * 32 * 2);

    auto g1 = [](int n) { return dim3((n + 255) / 256); };

    // ---- weight packing (deterministic, every call) ----
    packlin_k<<<g1(2048 * 160), 256, 0, stream>>>(l1_w, wlinT);
    packconv_k<<<g1(9 * 32 * 64), 256, 0, stream>>>(e_w2, wpEnc, 32, 32, 64);
    for (int i = 0; i < 3; ++i) {
        packconv_k<<<g1(9 * 64 * 96), 256, 0, stream>>>(cw1[i], wpC1[i], 64, 64, 96);
        packconv_k<<<g1(9 * 32 * 64), 256, 0, stream>>>(cw2[i], wpC2[i], 32, 32, 64);
    }
    packconv_k<<<g1(9 * 64 * 32),  256, 0, stream>>>(up1_w, wpU1, 64, 64, 32);
    packconv_k<<<g1(9 * 128 * 64), 256, 0, stream>>>(up2_w, wpU2, 128, 128, 64);
    packconv_k<<<g1(9 * 64 * 128), 256, 0, stream>>>(d1_w, wpD1, 64, 64, 128);
    packconv_k<<<g1(9 * 32 * 64),  256, 0, stream>>>(d2_w, wpD2, 32, 32, 64);
    packconv_k<<<g1(9 * 16 * 32),  256, 0, stream>>>(d3_w, wpD3, 1, 16, 32);

    // ---- linear path: zy @ l1_w -> init features (NHWC, reshaped) ----
    zy_k<<<g1(NROOMS * 160), 256, 0, stream>>>(z, gy, nd, zy);
    lin_gemm_k<<<1024, 256, 0, stream>>>(wlinT, zy, l1_b, initf);

    // ---- encoder: conv1 (scalar) + WMMA conv2, x = lrelu(conv2) + initf ----
    enc1a_k<<<g1(NROOMS * 64 * 64), 256, 0, stream>>>(gm, e_w1, e_b1, e1);
    conv3x3_wmma_k<64, 32, 8, 8, 8, 0, 1>
        <<<dim3(NROOMS, 1), 256, 0, stream>>>(e1, wpEnc, e_b2, x8, initf, nullptr);

    // ---- 3 message-passing rounds ----
    for (int i = 0; i < 3; ++i) {
        hipMemsetAsync(pos, 0, (size_t)NROOMS * 2048 * 4, stream);
        hipMemsetAsync(neg, 0, (size_t)NROOMS * 2048 * 4, stream);
        edge_k<<<NEDGES, 256, 0, stream>>>(gw, x8, pos, neg);
        comb_k<<<g1(NROOMS * 64 * 96), 256, 0, stream>>>(x8, pos, neg, comb);
        conv3x3_wmma_k<96, 64, 8, 8, 8, 0, 1>
            <<<dim3(NROOMS, 1), 256, 0, stream>>>(comb, wpC1[i], cb1[i], h1, nullptr, nullptr);
        conv3x3_wmma_k<64, 32, 8, 8, 8, 0, 1>     // residual: x = x + lrelu(conv)
            <<<dim3(NROOMS, 1), 256, 0, stream>>>(h1, wpC2[i], cb2[i], x8, x8, nullptr);
    }

    // ---- x += 0.01 * noise ----
    noise_k<<<g1(NROOMS * 64 * 32), 256, 0, stream>>>(x8, nz);

    // ---- up1: up2 + conv 32->64 @16x16, then instance-norm + lrelu ----
    conv3x3_wmma_k<32, 64, 16, 16, 16, 1, 0>
        <<<dim3(NROOMS, 1), 256, 0, stream>>>(x8, wpU1, up1_b, a16, nullptr, nullptr);
    inorm_k<64, 256><<<NROOMS, 64, 0, stream>>>(a16);

    // ---- up2: up2 + conv 64->128 @32x32, instance-norm + lrelu ----
    conv3x3_wmma_k<64, 128, 32, 32, 8, 1, 0>
        <<<dim3(NROOMS, 4), 256, 0, stream>>>(a16, wpU2, up2_b, a32a, nullptr, nullptr);
    inorm_k<128, 1024><<<NROOMS, 128, 0, stream>>>(a32a);

    // ---- decoder ----
    conv3x3_wmma_k<128, 64, 32, 32, 4, 0, 1>
        <<<dim3(NROOMS, 8), 256, 0, stream>>>(a32a, wpD1, d1_b, a32b, nullptr, nullptr);
    conv3x3_wmma_k<64, 32, 32, 32, 8, 0, 1>
        <<<dim3(NROOMS, 4), 256, 0, stream>>>(a32b, wpD2, d2_b, a32c, nullptr, nullptr);
    conv3x3_wmma_k<32, 16, 32, 32, 16, 0, 2>     // sigmoid, channel 0 -> f32 out
        <<<dim3(NROOMS, 2), 256, 0, stream>>>(a32c, wpD3, d3_b, nullptr, nullptr, outp);

    (void)in_sizes; (void)n_in; (void)out_size; (void)ws_size;
}
